// MoELayer_5935644803099
// MI455X (gfx1250) — matrix operand order repaired and checked by reading, not compiled, not run on hardware
//
#include <hip/hip_runtime.h>
#include <hip/hip_bf16.h>
#include <math.h>

// Problem dims (fixed by the reference)
#define BB 2
#define SS 2048
#define DD 1024
#define HH 2048
#define EE 8
#define NTOK (BB * SS)              // 4096 tokens
#define MOE_OUT_ELEMS (NTOK * DD)   // 4,194,304 f32
#define TILE_M 32                   // tokens per block
#define TILES_PER_E (NTOK / TILE_M) // 128 (worst case: all tokens -> one expert)

typedef __bf16 bf16_t;
typedef bf16_t v16bf __attribute__((ext_vector_type(16)));
typedef bf16_t v8bf  __attribute__((ext_vector_type(8)));
typedef float  v8f   __attribute__((ext_vector_type(8)));

// LDS row stride: 1032 bf16 = 2064 B -> 16B aligned, stride % 256B == 16
// so 16 concurrent fragment rows land on distinct bank groups.
#define A_STR 1032

#define WMMA_BF16(a, b, c) \
    __builtin_amdgcn_wmma_f32_16x16x32_bf16(false, (a), false, (b), (short)0, (c), false, false)

// ---------------------------------------------------------------------------
// Kernel 0: zero moe output region + per-expert counters
// ---------------------------------------------------------------------------
__global__ __launch_bounds__(256) void moe_zero_kernel(float* __restrict__ out,
                                                       int* __restrict__ cnt) {
    size_t i = (size_t)blockIdx.x * blockDim.x + threadIdx.x;   // one float4 each
    ((float4*)out)[i] = make_float4(0.f, 0.f, 0.f, 0.f);
    if (i < EE) cnt[i] = 0;
}

// ---------------------------------------------------------------------------
// Kernel 1: router. One wave32 per token.
// ---------------------------------------------------------------------------
__global__ __launch_bounds__(256) void moe_router_kernel(
    const float* __restrict__ X, const float* __restrict__ wr,
    float* __restrict__ probs_out,
    int* __restrict__ cnt, int* __restrict__ lst, float* __restrict__ gwbuf) {
    const int lane = threadIdx.x & 31;
    const int w    = threadIdx.x >> 5;
    const int t    = blockIdx.x * 8 + w;
    if (t >= NTOK) return;

    float acc[EE];
#pragma unroll
    for (int e = 0; e < EE; ++e) acc[e] = 0.f;

    const float* xr = X + (size_t)t * DD;
    for (int d = lane; d < DD; d += 32) {
        float xv = xr[d];
        const float* wrow = wr + (size_t)d * EE;   // wr is [D][E] row-major
#pragma unroll
        for (int e = 0; e < EE; ++e) acc[e] += xv * wrow[e];
    }
#pragma unroll
    for (int off = 16; off > 0; off >>= 1)
#pragma unroll
        for (int e = 0; e < EE; ++e) acc[e] += __shfl_xor(acc[e], off, 32);

    // double softmax (reference applies softmax twice)
#pragma unroll
    for (int pass = 0; pass < 2; ++pass) {
        float m = acc[0];
#pragma unroll
        for (int e = 1; e < EE; ++e) m = fmaxf(m, acc[e]);
        float s = 0.f;
#pragma unroll
        for (int e = 0; e < EE; ++e) { acc[e] = __expf(acc[e] - m); s += acc[e]; }
        float inv = __builtin_amdgcn_rcpf(s);
#pragma unroll
        for (int e = 0; e < EE; ++e) acc[e] *= inv;
    }
    int i1 = 0;
#pragma unroll
    for (int e = 1; e < EE; ++e) if (acc[e] > acc[i1]) i1 = e;
    int i2 = (i1 == 0) ? 1 : 0;
#pragma unroll
    for (int e = 0; e < EE; ++e) if (e != i1 && acc[e] > acc[i2]) i2 = e;

    if (lane == 0) {
#pragma unroll
        for (int e = 0; e < EE; ++e) probs_out[(size_t)t * EE + e] = acc[e];
        int p1 = atomicAdd(&cnt[i1], 1);
        lst[i1 * NTOK + p1] = t;  gwbuf[i1 * NTOK + p1] = acc[i1];
        int p2 = atomicAdd(&cnt[i2], 1);
        lst[i2 * NTOK + p2] = t;  gwbuf[i2 * NTOK + p2] = acc[i2];
    }
}

// ---------------------------------------------------------------------------
// Helpers
// ---------------------------------------------------------------------------
// A (16x32 bf16) fragment, ISA 7.12.2: lane L -> row m=L%16; two contiguous
// 16-byte runs at k offsets {0|8} and {16|24} depending on lane half.
__device__ inline v16bf load_a_frag(const bf16_t* row, int k0, int hi) {
    const bf16_t* p = row + k0 + (hi ? 8 : 0);
    v8bf lo  = *(const v8bf*)(p);
    v8bf hi8 = *(const v8bf*)(p + 16);
    return __builtin_shufflevector(lo, hi8, 0, 1, 2, 3, 4, 5, 6, 7,
                                   8, 9, 10, 11, 12, 13, 14, 15);
}
// B (32x16 bf16) fragment: lane L holds column n=L%16; element i is k=8*hi+i
// (i<8) or k=16+8*hi+(i-8) (i>=8), matching the A fragment order.
// Source f32 -> packed bf16 converts (v_cvt_pk_bf16_f32).
__device__ inline v16bf load_b_frag_f32(const float* p, int stride, int hi) {
    v16bf b;
#pragma unroll
    for (int i = 0; i < 16; ++i) {
        int k = (i < 8) ? (8 * hi + i) : (16 + 8 * hi + (i - 8));
        b[i] = (bf16_t)p[(size_t)k * stride];
    }
    return b;
}
// Branch-free exact-style GELU: Abramowitz-Stegun 7.1.26 erf (|err|~1.5e-7).
// v_rcp_f32 instead of IEEE divide (div_scale/div_fmas chain) -- inputs are
// already bf16-quantized so ~1 ulp rcp is below the noise floor.
__device__ inline float gelu_erf(float g) {
    float x  = g * 0.70710678118654752f;
    float ax = fabsf(x);
    float t  = __builtin_amdgcn_rcpf(1.0f + 0.3275911f * ax);
    float p  = ((((1.061405429f * t - 1.453152027f) * t + 1.421413741f) * t
                 - 0.284496736f) * t + 0.254829592f) * t;
    float y  = 1.0f - p * __expf(-x * x);
    float er = copysignf(y, x);
    return 0.5f * g * (1.0f + er);
}

// ---------------------------------------------------------------------------
// Kernel 2: fused expert FFN over gathered tokens.
//   Block = (expert e, 32-token tile), 256 threads = 8 waves.
//   H processed in 2 halves so the act tile is 32x1024 (66 KB); GEMM2 keeps
//   persistent f32 accumulators in VGPRs across halves (128 VGPRs/lane).
// ---------------------------------------------------------------------------
__global__ __launch_bounds__(256) void moe_ffn_kernel(
    const float* __restrict__ X, const float* __restrict__ w1,
    const float* __restrict__ w2, const int* __restrict__ cnt,
    const int* __restrict__ lst, const float* __restrict__ gwbuf,
    float* __restrict__ out) {
    __shared__ __attribute__((aligned(16))) bf16_t As[TILE_M * A_STR];   // 66 KB
    __shared__ __attribute__((aligned(16))) bf16_t Act[TILE_M * A_STR];  // 66 KB
    __shared__ int   tokS[TILE_M];
    __shared__ float gwS[TILE_M];

    const int e    = blockIdx.x / TILES_PER_E;
    const int tile = blockIdx.x % TILES_PER_E;
    const int n_e  = cnt[e];
    const int base = tile * TILE_M;
    if (base >= n_e) return;   // uniform exit: most tiles idle after routing

    const int tid  = threadIdx.x;
    const int lane = tid & 31;
    const int w    = tid >> 5;
    const int hi   = lane >> 4;
    const int mrow = lane & 15;
    const int nlan = lane & 15;

    if (tid < TILE_M) {
        int r = base + tid;
        if (r < n_e) { tokS[tid] = lst[e * NTOK + r]; gwS[tid] = gwbuf[e * NTOK + r]; }
        else         { tokS[tid] = 0;                 gwS[tid] = 0.f; }
    }
    __syncthreads();

    // Stage 32 token rows of X as bf16 into LDS
    for (int idx = tid; idx < TILE_M * DD; idx += 256) {
        int m = idx >> 10, d = idx & (DD - 1);
        float v = (gwS[m] != 0.f) ? X[(size_t)tokS[m] * DD + d] : 0.f;
        As[m * A_STR + d] = (bf16_t)v;
    }
    __syncthreads();

    const float* w1e = w1 + (size_t)e * DD * (2 * HH);
    const float* w2e = w2 + (size_t)e * HH * DD;

    // Persistent GEMM2 accumulators: 8 N-chunks x 2 M-tiles x v8f
    v8f acc2[8][2];
#pragma unroll
    for (int q = 0; q < 8; ++q) {
        acc2[q][0] = (v8f){0.f, 0.f, 0.f, 0.f, 0.f, 0.f, 0.f, 0.f};
        acc2[q][1] = (v8f){0.f, 0.f, 0.f, 0.f, 0.f, 0.f, 0.f, 0.f};
    }

    for (int half = 0; half < 2; ++half) {
        const int jbase = half * (HH / 2);

        // ---- GEMM1 for this H-half: act = x * gelu(gate), into LDS tile ----
        for (int c = w; c < (HH / 2) / 16; c += 8) {     // 64 chunks, 8/wave
            const int j = jbase + c * 16;
            v8f ax0 = {0.f,0.f,0.f,0.f,0.f,0.f,0.f,0.f};
            v8f ag0 = {0.f,0.f,0.f,0.f,0.f,0.f,0.f,0.f};
            v8f ax1 = {0.f,0.f,0.f,0.f,0.f,0.f,0.f,0.f};
            v8f ag1 = {0.f,0.f,0.f,0.f,0.f,0.f,0.f,0.f};
            for (int k0 = 0; k0 < DD; k0 += 32) {
                v16bf a0 = load_a_frag(&As[mrow * A_STR], k0, hi);
                v16bf a1 = load_a_frag(&As[(mrow + 16) * A_STR], k0, hi);
                const float* bxp = w1e + (size_t)k0 * (2 * HH) + j + nlan;
                // speculative prefetch of next K-step rows (global_prefetch_b8)
                __builtin_prefetch(bxp + 32 * (2 * HH), 0, 3);
                __builtin_prefetch(bxp + 32 * (2 * HH) + HH, 0, 3);
                v16bf bx = load_b_frag_f32(bxp, 2 * HH, hi);
                v16bf bg = load_b_frag_f32(bxp + HH, 2 * HH, hi);   // gate = h[:, H:]
                ax0 = WMMA_BF16(a0, bx, ax0);
                ag0 = WMMA_BF16(a0, bg, ag0);
                ax1 = WMMA_BF16(a1, bx, ax1);
                ag1 = WMMA_BF16(a1, bg, ag1);
            }
#pragma unroll
            for (int r = 0; r < 8; ++r) {
                int m0 = r + 8 * hi;
                Act[m0 * A_STR + c * 16 + nlan]        = (bf16_t)(gelu_erf(ag0[r]) * ax0[r]);
                Act[(m0 + 16) * A_STR + c * 16 + nlan] = (bf16_t)(gelu_erf(ag1[r]) * ax1[r]);
            }
        }
        __syncthreads();

        // ---- GEMM2 partial accumulate over this K-half ----
        // k-outer / q-inner: each LDS A-fragment pair feeds 16 WMMAs.
        for (int k0 = 0; k0 < HH / 2; k0 += 32) {
            v16bf a0 = load_a_frag(&Act[mrow * A_STR], k0, hi);
            v16bf a1 = load_a_frag(&Act[(mrow + 16) * A_STR], k0, hi);
            const float* brow = w2e + (size_t)(jbase + k0) * DD + w * 128 + nlan;
            __builtin_prefetch(brow + 32 * DD, 0, 3);   // next K-step row
#pragma unroll
            for (int q = 0; q < 8; ++q) {
                v16bf b = load_b_frag_f32(brow + q * 16, DD, hi);
                acc2[q][0] = WMMA_BF16(a0, b, acc2[q][0]);
                acc2[q][1] = WMMA_BF16(a1, b, acc2[q][1]);
            }
        }
        __syncthreads();   // Act reused by next half's GEMM1
    }

    // ---- Scale by gate weight and scatter (exactly 2 atomic adds/elem) ----
#pragma unroll
    for (int q = 0; q < 8; ++q) {
        const int n0 = w * 128 + q * 16;
#pragma unroll
        for (int r = 0; r < 8; ++r) {
            int m0 = r + 8 * hi;
            float g0 = gwS[m0];
            if (g0 != 0.f)
                unsafeAtomicAdd(&out[(size_t)tokS[m0] * DD + n0 + nlan], acc2[q][0][r] * g0);
            int m1 = m0 + 16;
            float g1 = gwS[m1];
            if (g1 != 0.f)
                unsafeAtomicAdd(&out[(size_t)tokS[m1] * DD + n0 + nlan], acc2[q][1][r] * g1);
        }
    }
}

// ---------------------------------------------------------------------------
extern "C" void kernel_launch(void* const* d_in, const int* in_sizes, int n_in,
                              void* d_out, int out_size, void* d_ws, size_t ws_size,
                              hipStream_t stream) {
    const float* X  = (const float*)d_in[0];   // [B,S,D]
    const float* w1 = (const float*)d_in[1];   // [E,D,2H]
    const float* w2 = (const float*)d_in[2];   // [E,H,D]
    const float* wr = (const float*)d_in[3];   // [D,E]
    float* out = (float*)d_out;                // [moe_out | routing_probs]

    char* ws = (char*)d_ws;
    int*   cnt   = (int*)ws;                                   // E counters
    int*   lst   = (int*)(ws + 256);                           // E * NTOK token ids
    float* gwbuf = (float*)(ws + 256 + (size_t)EE * NTOK * 4); // E * NTOK gate weights

    moe_zero_kernel<<<MOE_OUT_ELEMS / 4 / 256, 256, 0, stream>>>(out, cnt);
    moe_router_kernel<<<NTOK / 8, 256, 0, stream>>>(X, wr, out + MOE_OUT_ELEMS,
                                                    cnt, lst, gwbuf);
    moe_ffn_kernel<<<EE * TILES_PER_E, 256, 0, stream>>>(X, w1, w2, cnt, lst, gwbuf, out);
}
